// Attention_5403068858714
// MI455X (gfx1250) — matrix operand order repaired and checked by reading, not compiled, not run on hardware
//
#include <hip/hip_runtime.h>
#include <hip/hip_bf16.h>

#define B_   4
#define S_   2048
#define D_   1024
#define H_   16
#define HD_  64
#define WIN_ 256

typedef __attribute__((ext_vector_type(16))) _Float16 v16h;
typedef __attribute__((ext_vector_type(8)))  _Float16 h8;
typedef __attribute__((ext_vector_type(8)))  float    v8f;

// ---------------------------------------------------------------------------
// WMMA helper: D = A(16x32 f16) * B(32x16 f16) + C(16x16 f32)
// ---------------------------------------------------------------------------
__device__ __forceinline__ v8f wmma16(v16h a, v16h b, v8f c) {
  return __builtin_amdgcn_wmma_f32_16x16x32_f16(false, a, false, b, (short)0, c,
                                                false, false);
}

// A-matrix fragment (16x32, f16), row-major source with leading dim ldk.
// CDNA5 layout: lane L (r=L%16, p=L/16) holds row r, K = {8p..8p+7, 8p+16..8p+23}.
__device__ __forceinline__ v16h load_A16(const _Float16* base, int ldk, int m0,
                                         int k0, int lane) {
  const int p = lane >> 4, r = lane & 15;
  const _Float16* ptr = base + (size_t)(m0 + r) * (size_t)ldk + (k0 + 8 * p);
  h8 lo = *(const h8*)ptr;
  h8 hi = *(const h8*)(ptr + 16);
  v16h a;
#pragma unroll
  for (int i = 0; i < 8; ++i) { a[i] = lo[i]; a[i + 8] = hi[i]; }
  return a;
}

// B-matrix fragment (32x16, f16) where B(k,n) = base[(n0+n)*ldk + k0+k]
// i.e. computes A * Msub^T for row-major M.  Lane L (c=L%16, p=L/16)
// holds col c, K = {8p..8p+7, 8p+16..8p+23}.
__device__ __forceinline__ v16h load_B16(const _Float16* base, int ldk, int n0,
                                         int k0, int lane) {
  const int p = lane >> 4, c = lane & 15;
  const _Float16* ptr = base + (size_t)(n0 + c) * (size_t)ldk + (k0 + 8 * p);
  h8 lo = *(const h8*)ptr;
  h8 hi = *(const h8*)(ptr + 16);
  v16h b;
#pragma unroll
  for (int i = 0; i < 8; ++i) { b[i] = lo[i]; b[i + 8] = hi[i]; }
  return b;
}

// ---------------------------------------------------------------------------
// Kernel 0: f32 -> f16 conversion of x and weights (wq|wk|wv fused)
// ---------------------------------------------------------------------------
__global__ __launch_bounds__(256) void convert_kernel(
    const float* __restrict__ x, const float* __restrict__ wq,
    const float* __restrict__ wk, const float* __restrict__ wv,
    const float* __restrict__ wo, _Float16* __restrict__ xh,
    _Float16* __restrict__ wqkv, _Float16* __restrict__ woh) {
  const size_t NX = (size_t)B_ * S_ * D_;
  const size_t NW = (size_t)D_ * D_;
  const size_t total = NX + 4 * NW;
  for (size_t i = (size_t)blockIdx.x * blockDim.x + threadIdx.x; i < total;
       i += (size_t)gridDim.x * blockDim.x) {
    if (i < NX) {
      xh[i] = (_Float16)x[i];
    } else {
      size_t j = i - NX;
      if (j < NW)            wqkv[j]          = (_Float16)wq[j];
      else if (j < 2 * NW)   wqkv[j]          = (_Float16)wk[j - NW];
      else if (j < 3 * NW)   wqkv[j]          = (_Float16)wv[j - 2 * NW];
      else                   woh[j - 3 * NW]  = (_Float16)wo[j - 3 * NW];
    }
  }
}

// ---------------------------------------------------------------------------
// Kernel 1/4: register-blocked WMMA GEMM, out = A(MxK) * W(NxK)^T.
// One wave computes a 64x64 output supertile (4x4 fragments): per K-step
// 8 fragment loads feed 16 v_wmma -> 4x better wmma/byte than 1x1 tiling.
// ---------------------------------------------------------------------------
template <typename OutT>
__global__ __launch_bounds__(128) void gemm_wmma_blocked(
    const _Float16* __restrict__ A, const _Float16* __restrict__ W,
    OutT* __restrict__ out, int M, int N, int K) {
  const int lane = threadIdx.x & 31;
  const int wave = threadIdx.x >> 5;
  const int NT = N >> 6;
  const int MT = M >> 6;
  const int tile = blockIdx.x * (blockDim.x >> 5) + wave;
  if (tile >= MT * NT) return;
  const int mt = tile / NT, nt = tile % NT;
  const int m0 = mt << 6, n0 = nt << 6;

  v8f acc[4][4] = {};
  for (int k0 = 0; k0 < K; k0 += 32) {
    v16h af[4], bf[4];
#pragma unroll
    for (int i = 0; i < 4; ++i) af[i] = load_A16(A, K, m0 + 16 * i, k0, lane);
#pragma unroll
    for (int j = 0; j < 4; ++j) bf[j] = load_B16(W, K, n0 + 16 * j, k0, lane);
#pragma unroll
    for (int i = 0; i < 4; ++i)
#pragma unroll
      for (int j = 0; j < 4; ++j)
        acc[i][j] = wmma16(af[i], bf[j], acc[i][j]);
  }

  const int p = lane >> 4, c = lane & 15;
#pragma unroll
  for (int i = 0; i < 4; ++i)
#pragma unroll
    for (int j = 0; j < 4; ++j)
#pragma unroll
      for (int g = 0; g < 8; ++g) {
        const int row = m0 + 16 * i + g + 8 * p;
        out[(size_t)row * N + (n0 + 16 * j + c)] = (OutT)acc[i][j][g];
      }
}

// ---------------------------------------------------------------------------
// Kernel 2: RoPE + repack.  qkvh: [B*S][3072] f16 (q|k|v, head-major cols).
// Writes q,k as [B,H,S,64] (q scaled by 1/sqrt(64)), v transposed [B,H,64,S].
// ---------------------------------------------------------------------------
__global__ __launch_bounds__(256) void rope_pack_kernel(
    const _Float16* __restrict__ qkvh, const float* __restrict__ fcos,
    const float* __restrict__ fsin, _Float16* __restrict__ qh,
    _Float16* __restrict__ kh, _Float16* __restrict__ vT) {
  const int idx = blockIdx.x * blockDim.x + threadIdx.x;  // (b*H+h)*S + s
  if (idx >= B_ * H_ * S_) return;
  const int s = idx % S_;
  const int bh = idx / S_;
  const int h = bh % H_;
  const int b = bh / H_;

  const _Float16* qrow = qkvh + ((size_t)(b * S_ + s)) * (3 * D_) + h * HD_;
  const _Float16* krow = qrow + D_;
  const _Float16* vrow = qrow + 2 * D_;
  _Float16* qo = qh + ((size_t)bh * S_ + s) * HD_;
  _Float16* ko = kh + ((size_t)bh * S_ + s) * HD_;
  const float* cs = fcos + (size_t)s * (HD_ / 2);
  const float* sn = fsin + (size_t)s * (HD_ / 2);

#pragma unroll 4
  for (int i = 0; i < HD_ / 2; ++i) {
    const float c = cs[i], sv = sn[i];
    const float q0 = (float)qrow[2 * i], q1 = (float)qrow[2 * i + 1];
    const float k0 = (float)krow[2 * i], k1 = (float)krow[2 * i + 1];
    qo[2 * i]     = (_Float16)(0.125f * (q0 * c - q1 * sv));
    qo[2 * i + 1] = (_Float16)(0.125f * (q0 * sv + q1 * c));
    ko[2 * i]     = (_Float16)(k0 * c - k1 * sv);
    ko[2 * i + 1] = (_Float16)(k0 * sv + k1 * c);
  }
#pragma unroll 8
  for (int d = 0; d < HD_; ++d)
    vT[((size_t)bh * HD_ + d) * S_ + s] = vrow[d];
}

// ---------------------------------------------------------------------------
// Online-softmax update for one 32-key block.  MASKED=false is the interior
// fast path (all keys valid for all rows): no compares / cndmasks at all.
// ---------------------------------------------------------------------------
template <bool MASKED>
__device__ __forceinline__ void softmax_update(const v8f s[2], int kb, int m0,
                                               int p, int c, float* mst,
                                               float* lst, v8f acc[4],
                                               _Float16* Plds) {
#pragma unroll
  for (int g = 0; g < 8; ++g) {
    float a0, a1;
    bool v0 = true, v1 = true;
    const int row = m0 + g + 8 * p;
    if (MASKED) {
      const int key0 = kb + c, key1 = kb + 16 + c;
      v0 = (key0 <= row) && (row - key0 < WIN_);
      v1 = (key1 <= row) && (row - key1 < WIN_);
      a0 = v0 ? s[0][g] : -1e30f;
      a1 = v1 ? s[1][g] : -1e30f;
    } else {
      a0 = s[0][g];
      a1 = s[1][g];
    }
    float mx = fmaxf(a0, a1);
#pragma unroll
    for (int off = 1; off < 16; off <<= 1)
      mx = fmaxf(mx, __shfl_xor(mx, off, 32));  // stays inside half-wave
    const float mn = fmaxf(mst[g], mx);
    const float scale = __expf(mst[g] - mn);
    const float p0 = (!MASKED || v0) ? __expf(a0 - mn) : 0.f;
    const float p1 = (!MASKED || v1) ? __expf(a1 - mn) : 0.f;
    float rs = p0 + p1;
#pragma unroll
    for (int off = 1; off < 16; off <<= 1)
      rs += __shfl_xor(rs, off, 32);
    lst[g] = lst[g] * scale + rs;
    mst[g] = mn;
#pragma unroll
    for (int t = 0; t < 4; ++t) acc[t][g] *= scale;
    Plds[(g + 8 * p) * 32 + c] = (_Float16)p0;
    Plds[(g + 8 * p) * 32 + 16 + c] = (_Float16)p1;
  }
}

// ---------------------------------------------------------------------------
// Kernel 3: sliding-window causal flash attention.  One wave (32 threads) per
// (b, h, 16-query tile).  Scores/PV via WMMA, online softmax via shuffles,
// P re-layouted through LDS.
// ---------------------------------------------------------------------------
__global__ __launch_bounds__(32) void attn_kernel(
    const _Float16* __restrict__ qh, const _Float16* __restrict__ kh,
    const _Float16* __restrict__ vT, _Float16* __restrict__ ctx) {
  __shared__ _Float16 Plds[16 * 32];

  const int lane = threadIdx.x;
  const int qtiles = S_ / 16;
  const int qt = blockIdx.x % qtiles;
  const int bh = blockIdx.x / qtiles;
  const int h = bh % H_;
  const int b = bh / H_;
  const int m0 = qt * 16;

  const _Float16* qbase = qh + (size_t)bh * S_ * HD_;
  const _Float16* kbase = kh + (size_t)bh * S_ * HD_;
  const _Float16* vbase = vT + (size_t)bh * HD_ * S_;

  const v16h aq0 = load_A16(qbase, HD_, m0, 0, lane);
  const v16h aq1 = load_A16(qbase, HD_, m0, 32, lane);

  const int p = lane >> 4, c = lane & 15;
  float mst[8], lst[8];
  v8f acc[4] = {v8f{}, v8f{}, v8f{}, v8f{}};
#pragma unroll
  for (int g = 0; g < 8; ++g) { mst[g] = -1e30f; lst[g] = 0.f; }

  int lo = m0 - (WIN_ - 1);
  if (lo < 0) lo = 0;
  const int kb0 = lo & ~31;

  for (int kb = kb0; kb <= m0 + 15; kb += 32) {
    // ---- scores for 32 keys (two 16-col tiles), K = HD = 64 (2 wmma each)
    v8f s[2];
#pragma unroll
    for (int t = 0; t < 2; ++t) {
      v8f z = {};
      z = wmma16(aq0, load_B16(kbase, HD_, kb + 16 * t, 0, lane), z);
      z = wmma16(aq1, load_B16(kbase, HD_, kb + 16 * t, 32, lane), z);
      s[t] = z;
    }
    // ---- online softmax: interior blocks need no masking (uniform branch)
    const bool interior = (kb + 31 <= m0) && (kb >= m0 + 16 - WIN_);
    if (interior)
      softmax_update<false>(s, kb, m0, p, c, mst, lst, acc, Plds);
    else
      softmax_update<true>(s, kb, m0, p, c, mst, lst, acc, Plds);
    __syncthreads();
    // ---- PV: P(16x32) * V(32x16) for each of 4 output d-tiles
    const v16h aP = load_A16(&Plds[0], 32, 0, 0, lane);
#pragma unroll
    for (int t = 0; t < 4; ++t)
      acc[t] = wmma16(aP, load_B16(vbase, S_, 16 * t, kb, lane), acc[t]);
    __syncthreads();
  }

  // ---- normalize + write context [B,S,H*HD] f16
#pragma unroll
  for (int g = 0; g < 8; ++g) {
    const int row = m0 + g + 8 * p;
    const float inv = 1.0f / lst[g];
#pragma unroll
    for (int t = 0; t < 4; ++t)
      ctx[((size_t)(b * S_ + row)) * D_ + h * HD_ + 16 * t + c] =
          (_Float16)(acc[t][g] * inv);
  }
}

// ---------------------------------------------------------------------------
extern "C" void kernel_launch(void* const* d_in, const int* in_sizes, int n_in,
                              void* d_out, int out_size, void* d_ws,
                              size_t ws_size, hipStream_t stream) {
  const float* x    = (const float*)d_in[0];
  const float* fcos = (const float*)d_in[1];
  const float* fsin = (const float*)d_in[2];
  const float* wq   = (const float*)d_in[3];
  const float* wk   = (const float*)d_in[4];
  const float* wv   = (const float*)d_in[5];
  const float* wo   = (const float*)d_in[6];
  float* out = (float*)d_out;

  char* ws = (char*)d_ws;
  size_t off = 0;
  auto take = [&](size_t bytes) -> char* {
    char* pp = ws + off;
    off = (off + bytes + 255) & ~(size_t)255;
    return pp;
  };
  const size_t BS = (size_t)B_ * S_;
  _Float16* xh   = (_Float16*)take(BS * D_ * 2);           // x in f16
  _Float16* wqkv = (_Float16*)take((size_t)3 * D_ * D_ * 2);
  _Float16* woh  = (_Float16*)take((size_t)D_ * D_ * 2);
  _Float16* qkvh = (_Float16*)take(BS * 3 * D_ * 2);       // fused QKV result
  _Float16* qatt = (_Float16*)take(BS * D_ * 2);           // [B,H,S,64]
  _Float16* katt = (_Float16*)take(BS * D_ * 2);           // [B,H,S,64]
  _Float16* vatt = (_Float16*)take(BS * D_ * 2);           // [B,H,64,S]
  _Float16* ctx  = (_Float16*)take(BS * D_ * 2);           // [B,S,1024]

  // 0) convert to f16
  convert_kernel<<<2048, 256, 0, stream>>>(x, wq, wk, wv, wo, xh, wqkv, woh);

  // 1) fused QKV projection: [8192x1024] * [3072x1024]^T (64x64 per wave)
  {
    const int tiles = (B_ * S_ / 64) * (3 * D_ / 64);      // 6144 waves
    gemm_wmma_blocked<_Float16><<<(tiles + 3) / 4, 128, 0, stream>>>(
        xh, wqkv, qkvh, B_ * S_, 3 * D_, D_);
  }

  // 2) RoPE + head-major repack (+ V transpose)
  rope_pack_kernel<<<(B_ * H_ * S_ + 255) / 256, 256, 0, stream>>>(
      qkvh, fcos, fsin, qatt, katt, vatt);

  // 3) sliding-window flash attention, one wave per (b,h,qtile)
  attn_kernel<<<B_ * H_ * (S_ / 16), 32, 0, stream>>>(qatt, katt, vatt, ctx);

  // 4) output projection: [8192x1024] * [1024x1024]^T -> f32 out
  {
    const int tiles = (B_ * S_ / 64) * (D_ / 64);          // 2048 waves
    gemm_wmma_blocked<float><<<(tiles + 3) / 4, 128, 0, stream>>>(
        ctx, woh, out, B_ * S_, D_, D_);
  }
}
